// SSM_47141561041352
// MI455X (gfx1250) — matrix operand order repaired and checked by reading, not compiled, not run on hardware
//
#include <hip/hip_runtime.h>
#include <hip/hip_bf16.h>

// ---------------------------------------------------------------------------
// CDNA5 (gfx1250) SS2D forward.  wave32; WMMA f32<-f16 GEMMs; A tile staged by
// global_load_async_to_lds_b128 (ASYNCcnt), B tile staged by the Tensor Data
// Mover (tensor_load_to_lds, TENSORcnt); lane-parallel selective scan.
// ---------------------------------------------------------------------------

typedef __attribute__((ext_vector_type(16))) _Float16 v16h;
typedef __attribute__((ext_vector_type(8)))  float    v8f;
typedef __attribute__((ext_vector_type(4)))  unsigned u32x4;
typedef __attribute__((ext_vector_type(8)))  unsigned u32x8;

#define BB   8
#define DM   96
#define DI   192
#define NS   16
#define HW   1024
#define LSEQ 4096

// B tile lands in LDS with TDM padding: 1 DWORD pad per 128 DWORDs -> row
// stride 129 floats (bank-conflict-free column reads: 16*129 mod 64 = 16).
#define BROW 129

// ---------------------------------------------------------------------------
// Batched GEMM:  C[b] = act( W(MxK) * X[b](KxN) + bias )
// Contract: K % 32 == 0, N % 128 == 0 (all call sites satisfy this).
// Block = 128 threads (4 wave32). Block tile = 16M x 128N; each wave owns a
// 16x32 C tile (one A fragment feeds two v_wmma per k-step).
// ---------------------------------------------------------------------------
template <bool HAS_BIAS, int ACT>
__global__ __launch_bounds__(128) void ssm_gemm_wmma(
    const float* __restrict__ W, const float* __restrict__ X,
    const float* __restrict__ bias, float* __restrict__ C,
    int M, int K, int N,
    long long xBs, int ldx,
    long long cBs, int ldc)
{
  __shared__ __align__(16) float As[16][36];        // f32 A tile, padded rows
  __shared__ __align__(16) float Bk[32 * BROW];     // f32 B tile via TDM

  const int b   = blockIdx.z;
  const int m0  = blockIdx.y * 16;
  const int n0  = blockIdx.x * 128;
  const int tid = threadIdx.x;
  const int ln  = tid & 31;
  const int wv  = tid >> 5;
  const float* Xb = X + (long long)b * xBs;

  // A async-stage assignment: thread -> (row, 4-float segment)
  const int ar = tid >> 3;           // 0..15
  const int as = tid & 7;            // 0..7
  int gr = m0 + ar; if (gr > M - 1) gr = M - 1;       // clamp (x_proj M=38)
  const bool aPad = (m0 + ar) >= M;                   // row must read as zero
  const float* aRow = W + (long long)gr * K + as * 4;
  const unsigned aLds =
      (unsigned)(unsigned long long)(uintptr_t)&As[ar][as * 4];
  const unsigned bLds =
      (unsigned)(unsigned long long)(uintptr_t)&Bk[0];

  // TDM D# group1: invariant across k-steps.
  // dw0: data_size=4B (2<<16) | pad_enable (1<<20) | pad_interval=6 (128 dw)
  //      | pad_amount=0 (1 dw); workgroup_mask = 0 (not in a cluster)
  u32x8 g1;
  g1[0] = (2u << 16) | (1u << 20) | (6u << 22);
  g1[1] = ((unsigned)ldx & 0xFFFFu) << 16;                    // tensor_dim0 lo
  g1[2] = (((unsigned)ldx >> 16) & 0xFFFFu) |
          (((unsigned)K & 0xFFFFu) << 16);                    // dim0 hi | dim1 lo
  g1[3] = (((unsigned)K >> 16) & 0xFFFFu) | (128u << 16);     // dim1 hi | tile0
  g1[4] = 32u;                                                // tile1=32, tile2=0
  g1[5] = (unsigned)ldx;                                      // dim0_stride lo
  g1[6] = 0u;                                                 // stride hi | dim1_stride lo
  g1[7] = 0u;

  const int row = ln & 15;
  const int hi  = ln >> 4;

  v8f acc0 = {}, acc1 = {};

  for (int k0 = 0; k0 < K; k0 += 32) {
    // ---- A tile: one async b128 per thread, global -> LDS, raw f32 ----
    {
      unsigned long long ga = (unsigned long long)(aRow + k0);
      asm volatile("global_load_async_to_lds_b128 %0, %1, off"
                   :: "v"(aLds), "v"(ga) : "memory");
    }
    // ---- B tile: one TDM descriptor per block (wave 0 issues it) ----
    if (wv == 0) {
      unsigned long long gb =
          (unsigned long long)(Xb + (long long)k0 * ldx + n0);
      u32x4 g0;
      g0[0] = 1u;                                   // count=1, user mode
      g0[1] = bLds;                                 // lds_addr
      g0[2] = (unsigned)gb;                         // global_addr[31:0]
      g0[3] = (unsigned)((gb >> 32) & 0x1FFFFFFu) | (2u << 30);  // addr hi|type=2
      asm volatile("tensor_load_to_lds %0, %1"
                   :: "s"(g0), "s"(g1) : "memory");
    }
    __builtin_amdgcn_s_wait_tensorcnt(0);
    asm volatile("s_wait_asynccnt 0x0" ::: "memory");
    // zero the (rare) OOB A rows after the async load landed
    if (aPad) {
      As[ar][as * 4 + 0] = 0.f; As[ar][as * 4 + 1] = 0.f;
      As[ar][as * 4 + 2] = 0.f; As[ar][as * 4 + 3] = 0.f;
    }
    __syncthreads();

    // ---- fragments (f32 in LDS -> f16 at read) ----
    v16h af, bf0, bf1;
#pragma unroll
    for (int j = 0; j < 16; ++j) {
      // A layout: lane = M row, element j -> K = 16*hi + j
      int ka = hi * 16 + j;
      af[j] = (_Float16)As[row][ka];
      // B layout: lane = N column, element j -> K = 16*hi + j
      int kb = hi * 16 + j;
      bf0[j] = (_Float16)Bk[kb * BROW + wv * 32 + row];
      bf1[j] = (_Float16)Bk[kb * BROW + wv * 32 + 16 + row];
    }
    acc0 = __builtin_amdgcn_wmma_f32_16x16x32_f16(
        false, af, false, bf0, (short)0, acc0, false, false);
    acc1 = __builtin_amdgcn_wmma_f32_16x16x32_f16(
        false, af, false, bf1, (short)0, acc1, false, false);
    __syncthreads();
  }

  // ---- store: VGPR i -> M = m0 + i + 8*hi ; cols wv*32+row, +16 ----
  float* Cb = C + (long long)b * cBs;
  const int col0 = n0 + wv * 32 + row;
  const int col1 = col0 + 16;
#pragma unroll
  for (int i = 0; i < 8; ++i) {
    int m = m0 + i + hi * 8;
    if (m < M) {
      float v0 = acc0[i], v1 = acc1[i];
      if constexpr (HAS_BIAS) { float bv = bias[m]; v0 += bv; v1 += bv; }
      if constexpr (ACT == 1) { v0 = fmaxf(v0, 0.f); v1 = fmaxf(v1, 0.f); }
      Cb[(long long)m * ldc + col0] = v0;
      Cb[(long long)m * ldc + col1] = v1;
    }
  }
}

// ---------------------------------------------------------------------------
// Depthwise 3x3 conv, SAME zero padding, 32x32 spatial, 192 channels, B=8.
// act: 0 none, 1 relu, 2 silu
// ---------------------------------------------------------------------------
__global__ __launch_bounds__(256) void ssm_dwconv3x3(
    const float* __restrict__ in, long long inBs, int inDs,
    const float* __restrict__ w, const float* __restrict__ bias,
    float* __restrict__ out, long long outBs, int outDs, int act)
{
  int idx = blockIdx.x * 256 + threadIdx.x;
  if (idx >= BB * DI * HW) return;
  int hw = idx & (HW - 1);
  int d  = (idx >> 10) % DI;
  int b  = idx / (DI * HW);
  int yy = hw >> 5, xx = hw & 31;
  const float* ip = in + (long long)b * inBs + (long long)d * inDs;
  const float* wp = w + d * 9;
  float acc = 0.f;
#pragma unroll
  for (int ky = 0; ky < 3; ++ky) {
    int iy = yy + ky - 1;
    if (iy < 0 || iy > 31) continue;
#pragma unroll
    for (int kx = 0; kx < 3; ++kx) {
      int ix = xx + kx - 1;
      if (ix < 0 || ix > 31) continue;
      acc += wp[ky * 3 + kx] * ip[iy * 32 + ix];
    }
  }
  if (bias) acc += bias[d];
  if (act == 1) acc = fmaxf(acc, 0.f);
  else if (act == 2) acc = acc / (1.f + __expf(-acc));
  out[(long long)b * outBs + (long long)d * outDs + hw] = acc;
}

// ---------------------------------------------------------------------------
// delta = softplus(dt_proj(dt_rank rows of x_dbl) + 2*dt_proj_b)
// (reference adds the bias twice -- replicated faithfully)
// ---------------------------------------------------------------------------
__global__ __launch_bounds__(256) void ssm_dt_delta(
    const float* __restrict__ xdbl,   // (8,38,4096)
    const float* __restrict__ dtw,    // (192,6)
    const float* __restrict__ dtb,    // (192)
    float* __restrict__ delta)        // (8,192,4096)
{
  int idx = blockIdx.x * 256 + threadIdx.x;
  if (idx >= BB * DI * LSEQ) return;
  int l = idx & (LSEQ - 1);
  int d = (idx >> 12) % DI;
  int b = idx / (DI * LSEQ);
  const float* xp = xdbl + (long long)b * 38 * LSEQ;
  float acc = 2.f * dtb[d];
#pragma unroll
  for (int r = 0; r < 6; ++r)
    acc += xp[r * LSEQ + l] * dtw[d * 6 + r];
  float s = (acc > 20.f) ? acc : __logf(1.f + __expf(acc));
  delta[idx] = s;
}

// ---------------------------------------------------------------------------
// Transpose B/C rows of x_dbl into (b, L, 32) so the scan's 16-lane state
// groups read contiguous 128B per step.
// ---------------------------------------------------------------------------
__global__ __launch_bounds__(256) void ssm_bct(
    const float* __restrict__ xdbl,   // (8,38,4096)
    float* __restrict__ bct)          // (8,4096,32): [0..15]=B, [16..31]=C
{
  int idx = blockIdx.x * 256 + threadIdx.x;
  if (idx >= BB * LSEQ * 32) return;
  int j = idx & 31;
  int l = (idx >> 5) & (LSEQ - 1);
  int b = idx >> 17;
  bct[idx] = xdbl[(long long)b * 38 * LSEQ + (long long)(6 + j) * LSEQ + l];
}

// ---------------------------------------------------------------------------
// Selective scan: one lane per (b, d, n) state -> 24576 lanes = 768 wave32.
// Per step: v_exp_f32 + fma, then 4-hop shfl_xor reduction within the 16-lane
// state group; lane n==0 writes y.
// ---------------------------------------------------------------------------
__global__ __launch_bounds__(256) void ssm_scan(
    const float* __restrict__ xs,     // (8,192,4096)  u
    const float* __restrict__ delta,  // (8,192,4096)
    const float* __restrict__ bct,    // (8,4096,32)
    const float* __restrict__ A_log,  // (192,16)
    const float* __restrict__ Dp,     // (192)
    float* __restrict__ y)            // (8,192,4096)
{
  int tid = blockIdx.x * 256 + threadIdx.x;
  int chain = tid >> 4;               // b*192 + d
  int n = tid & 15;
  if (chain >= BB * DI) return;
  int b = chain / DI, d = chain - b * DI;

  float An = -__expf(A_log[d * NS + n]);
  float Dd = Dp[d];
  const float* u_p  = xs    + (long long)chain * LSEQ;
  const float* dl_p = delta + (long long)chain * LSEQ;
  const float* bc_p = bct   + (long long)b * LSEQ * 32;
  float* y_p        = y     + (long long)chain * LSEQ;

  float h = 0.f;
  for (int t = 0; t < LSEQ; ++t) {
    __builtin_prefetch(bc_p + (t + 16) * 32, 0, 0);   // global_prefetch_b8
    float u  = u_p[t];
    float dl = dl_p[t];
    float Bn = bc_p[t * 32 + n];
    float Cn = bc_p[t * 32 + 16 + n];
    h = __expf(dl * An) * h + (dl * u) * Bn;
    float p = h * Cn;
    p += __shfl_xor(p, 1, 32);
    p += __shfl_xor(p, 2, 32);
    p += __shfl_xor(p, 4, 32);
    p += __shfl_xor(p, 8, 32);
    if (n == 0) y_p[t] = p + Dd * u;
  }
}

// ---------------------------------------------------------------------------
// Softmax-weighted combine of the 4 scan segments, channel LayerNorm,
// gate with silu(z), producing yz for out_proj.
// ---------------------------------------------------------------------------
__global__ __launch_bounds__(256) void ssm_combine_ln(
    const float* __restrict__ y,      // (8,192,4096)
    const float* __restrict__ sw,     // (4)
    const float* __restrict__ lnw, const float* __restrict__ lnb,
    const float* __restrict__ xz,     // (8,384,1024); z rows 192..383 (raw)
    float* __restrict__ yc,           // (8,192,1024) scratch
    float* __restrict__ yz)           // (8,192,1024)
{
  int idx = blockIdx.x * 256 + threadIdx.x;
  if (idx >= BB * HW) return;
  int hw = idx & (HW - 1);
  int b  = idx >> 10;

  float e0 = __expf(sw[0]), e1 = __expf(sw[1]), e2 = __expf(sw[2]), e3 = __expf(sw[3]);
  float inv = 1.f / (e0 + e1 + e2 + e3);
  float w0 = e0 * inv, w1 = e1 * inv, w2 = e2 * inv, w3 = e3 * inv;

  float s = 0.f, s2 = 0.f;
  for (int d = 0; d < DI; ++d) {
    const float* yp = y + ((long long)b * DI + d) * LSEQ + hw;
    float v = w0 * yp[0] + w1 * yp[1024] + w2 * yp[2048] + w3 * yp[3072];
    yc[((long long)b * DI + d) * HW + hw] = v;
    s += v; s2 += v * v;
  }
  float mu  = s * (1.f / DI);
  float var = s2 * (1.f / DI) - mu * mu;
  float rs  = rsqrtf(var + 1e-5f);
  for (int d = 0; d < DI; ++d) {
    float v  = yc[((long long)b * DI + d) * HW + hw];
    float yn = (v - mu) * rs * lnw[d] + lnb[d];
    float zr = xz[(long long)b * 384 * HW + (long long)(DI + d) * HW + hw];
    float zs = zr / (1.f + __expf(-zr));
    yz[((long long)b * DI + d) * HW + hw] = yn * zs;
  }
}

// ---------------------------------------------------------------------------
// Host side
// ---------------------------------------------------------------------------
extern "C" void kernel_launch(void* const* d_in, const int* in_sizes, int n_in,
                              void* d_out, int out_size, void* d_ws, size_t ws_size,
                              hipStream_t stream) {
  const float* x         = (const float*)d_in[0];
  const float* in_proj_w = (const float*)d_in[1];   // (384,96)
  const float* conv_w    = (const float*)d_in[2];
  const float* conv_b    = (const float*)d_in[3];
  const float* x_proj_w  = (const float*)d_in[4];   // (38,192)
  const float* dt_proj_w = (const float*)d_in[5];   // (192,6)
  const float* dt_proj_b = (const float*)d_in[6];
  const float* A_log     = (const float*)d_in[7];
  const float* Dp        = (const float*)d_in[8];
  const float* scan_w    = (const float*)d_in[9];
  const float* ln_w      = (const float*)d_in[10];
  const float* ln_b      = (const float*)d_in[11];
  const float* out_pw    = (const float*)d_in[12];  // (96,192)
  const float* fc_dw_w   = (const float*)d_in[13];
  const float* fc_dw_b   = (const float*)d_in[14];
  const float* fc_pw_w   = (const float*)d_in[15];  // (192,192)
  const float* fc_pw_b   = (const float*)d_in[16];
  const float* fv_w      = (const float*)d_in[17];
  const float* fv_b      = (const float*)d_in[18];
  const float* fh_w      = (const float*)d_in[19];
  const float* fh_b      = (const float*)d_in[20];
  const float* sob_x     = (const float*)d_in[21];
  const float* sob_y     = (const float*)d_in[22];
  float* out = (float*)d_out;

  // workspace layout (floats)
  float* ws = (float*)d_ws;
  float* XZ    = ws;                         // (8,384,1024)  = 3,145,728
  float* XS    = XZ    + 3145728;            // (8,192,4096)  = 6,291,456
  float* TMP   = XS    + 6291456;            // (8,192,1024)  = 1,572,864
  float* XDBL  = TMP   + 1572864;            // (8,38,4096)   = 1,245,184
  float* DELTA = XDBL  + 1245184;            // (8,192,4096)  = 6,291,456
  float* BCT   = DELTA + 6291456;            // (8,4096,32)   = 1,048,576
  float* Y     = BCT   + 1048576;            // (8,192,4096)  = 6,291,456
  float* YC    = Y     + 6291456;            // (8,192,1024)  = 1,572,864
  float* YZ    = YC    + 1572864;            // (8,192,1024)  = 1,572,864
  (void)ws_size; (void)n_in; (void)in_sizes; (void)out_size;

  const long long xzBs = 384LL * HW, xsBs = (long long)DI * LSEQ;
  const long long tpBs = (long long)DI * HW;

  // 1. in_proj: XZ = W(384x96) * x[b](96x1024)
  ssm_gemm_wmma<false, 0><<<dim3(HW / 128, 384 / 16, BB), 128, 0, stream>>>(
      in_proj_w, x, nullptr, XZ, 384, DM, HW,
      (long long)DM * HW, HW, xzBs, HW);

  // 2. xm = silu(dwconv3x3(XZ rows 0..191) + conv_b) -> XS segment 0
  const int ew = (BB * DI * HW + 255) / 256;
  ssm_dwconv3x3<<<ew, 256, 0, stream>>>(XZ, xzBs, HW, conv_w, conv_b,
                                        XS, xsBs, LSEQ, 2);

  // 3. vert branch: tmp = sobel_x(xm); XS seg1 = relu(dwconv(tmp, fv)+fv_b)
  ssm_dwconv3x3<<<ew, 256, 0, stream>>>(XS, xsBs, LSEQ, sob_x, nullptr,
                                        TMP, tpBs, HW, 0);
  ssm_dwconv3x3<<<ew, 256, 0, stream>>>(TMP, tpBs, HW, fv_w, fv_b,
                                        XS + 1024, xsBs, LSEQ, 1);

  // 4. auto branch: tmp = dwconv(xm, fc_dw)+b ; XS seg2 = relu(pw * tmp + b)
  ssm_dwconv3x3<<<ew, 256, 0, stream>>>(XS, xsBs, LSEQ, fc_dw_w, fc_dw_b,
                                        TMP, tpBs, HW, 0);
  ssm_gemm_wmma<true, 1><<<dim3(HW / 128, DI / 16, BB), 128, 0, stream>>>(
      fc_pw_w, TMP, fc_pw_b, XS + 2048, DI, DI, HW,
      tpBs, HW, xsBs, LSEQ);

  // 5. hori branch
  ssm_dwconv3x3<<<ew, 256, 0, stream>>>(XS, xsBs, LSEQ, sob_y, nullptr,
                                        TMP, tpBs, HW, 0);
  ssm_dwconv3x3<<<ew, 256, 0, stream>>>(TMP, tpBs, HW, fh_w, fh_b,
                                        XS + 3072, xsBs, LSEQ, 1);

  // 6. x_proj: XDBL = W(38x192) * XS[b](192x4096)   (M=38, tiles guarded)
  ssm_gemm_wmma<false, 0><<<dim3(LSEQ / 128, 3, BB), 128, 0, stream>>>(
      x_proj_w, XS, nullptr, XDBL, 38, DI, LSEQ,
      xsBs, LSEQ, 38LL * LSEQ, LSEQ);

  // 7. delta = softplus(dt_proj + 2*bias)
  ssm_dt_delta<<<(BB * DI * LSEQ + 255) / 256, 256, 0, stream>>>(
      XDBL, dt_proj_w, dt_proj_b, DELTA);

  // 8. transpose B/C into (b, L, 32)
  ssm_bct<<<(BB * LSEQ * 32 + 255) / 256, 256, 0, stream>>>(XDBL, BCT);

  // 9. selective scan
  ssm_scan<<<(BB * DI * NS + 255) / 256, 256, 0, stream>>>(
      XS, DELTA, BCT, A_log, Dp, Y);

  // 10. combine + layernorm + silu(z) gate
  ssm_combine_ln<<<(BB * HW + 255) / 256, 256, 0, stream>>>(
      Y, scan_w, ln_w, ln_b, XZ, YC, YZ);

  // 11. out_proj -> d_out (8,96,32,32)
  ssm_gemm_wmma<false, 0><<<dim3(HW / 128, DM / 16, BB), 128, 0, stream>>>(
      out_pw, YZ, nullptr, out, DM, DI, HW,
      tpBs, HW, (long long)DM * HW, HW);
}
